// CosDistance_2594160247418
// MI455X (gfx1250) — matrix operand order repaired and checked by reading, not compiled, not run on hardware
//
#include <hip/hip_runtime.h>
#include <hip/hip_bf16.h>

// ---------------------------------------------------------------------------
// Cosine similarity: out[q,n] = dot(Q[q], S[n]) / max(||Q[q]||*||S[n]||, eps)
// Strategy: split-bf16 (hi+lo) WMMA GEMM, fp32 accumulate. Whole bf16 operand
// set (~106 MB) is L2-resident (192 MB); output (410 MB) streams with NT
// stores so it cannot evict the operands.
// ---------------------------------------------------------------------------

typedef __attribute__((ext_vector_type(16))) __bf16 v16bf;
typedef __attribute__((ext_vector_type(8)))  float  v8f;
typedef __attribute__((ext_vector_type(4)))  float  f4v;   // clang vector: OK for NT builtins
typedef __attribute__((ext_vector_type(2)))  float  f2v;   // clang vector: OK for NT builtins

#define NQ  2048
#define NS  50000
#define KD  512
#define EPSV 1e-8f

// ---------------------------------------------------------------------------
// Prep 1: support rows -> bf16 hi/lo (row-major) + row norms sn[NS]
// one block per row, 256 threads, 2 elements/thread. Source fp32 is read
// exactly once -> non-temporal loads keep it out of L2.
// ---------------------------------------------------------------------------
__global__ __launch_bounds__(256) void cos_prep_support(
    const float* __restrict__ S,
    __bf16* __restrict__ Shi, __bf16* __restrict__ Slo,
    float* __restrict__ sn)
{
    __shared__ float red[256];
    const int row = blockIdx.x;
    const int t   = threadIdx.x;
    const float* sr = S + (size_t)row * KD;

    f2v xv = __builtin_nontemporal_load((const f2v*)(sr + 2 * t));
    float ss = xv.x * xv.x + xv.y * xv.y;

    __bf16 h0 = (__bf16)xv.x;
    __bf16 h1 = (__bf16)xv.y;
    __bf16 l0 = (__bf16)(xv.x - (float)h0);
    __bf16 l1 = (__bf16)(xv.y - (float)h1);

    union { __bf16 b[2]; unsigned u; } ph, pl;
    ph.b[0] = h0; ph.b[1] = h1;
    pl.b[0] = l0; pl.b[1] = l1;
    *(unsigned*)(Shi + (size_t)row * KD + 2 * t) = ph.u;
    *(unsigned*)(Slo + (size_t)row * KD + 2 * t) = pl.u;

    red[t] = ss;
    __syncthreads();
    for (int ofs = 128; ofs > 0; ofs >>= 1) {
        if (t < ofs) red[t] += red[t + ofs];
        __syncthreads();
    }
    if (t == 0) sn[row] = sqrtf(red[0]);
}

// ---------------------------------------------------------------------------
// Prep 2: query rows -> row-major bf16 hi/lo Q[q][k] (2048 x 512) + qn[NQ]
// one block per query row. Row-contiguous 2B writes, 4 MB total.
// ---------------------------------------------------------------------------
__global__ __launch_bounds__(256) void cos_prep_query(
    const float* __restrict__ Q,
    __bf16* __restrict__ Qthi, __bf16* __restrict__ Qtlo,
    float* __restrict__ qn)
{
    __shared__ float red[256];
    const int q = blockIdx.x;
    const int t = threadIdx.x;
    const float* qr = Q + (size_t)q * KD;

    f2v xv = __builtin_nontemporal_load((const f2v*)(qr + 2 * t));
    float ss = xv.x * xv.x + xv.y * xv.y;

    __bf16 h0 = (__bf16)xv.x;
    __bf16 h1 = (__bf16)xv.y;
    __bf16 l0 = (__bf16)(xv.x - (float)h0);
    __bf16 l1 = (__bf16)(xv.y - (float)h1);

    const int k0 = 2 * t;
    Qthi[(size_t)q * KD + k0    ] = h0;
    Qthi[(size_t)q * KD + k0 + 1] = h1;
    Qtlo[(size_t)q * KD + k0    ] = l0;
    Qtlo[(size_t)q * KD + k0 + 1] = l1;

    red[t] = ss;
    __syncthreads();
    for (int ofs = 128; ofs > 0; ofs >>= 1) {
        if (t < ofs) red[t] += red[t + ofs];
        __syncthreads();
    }
    if (t == 0) qn[q] = sqrtf(red[0]);
}

// ---------------------------------------------------------------------------
// Main GEMM: A = support (M = n), B = Qt (N = q), D = 16x16 fp32 tiles.
// Workgroup: 256 threads = 8 waves, all stacked along M.
//   Each wave: 4 M-subtiles x 2 N-subtiles = 64 n x 32 q of output.
//   WG block : 512 n x 32 q. Grid: ceil(50000/512)=98 x (2048/32)=64.
// K-loop fully unrolled: all operand addresses become base + immediate
// IOFFSET (max ~17 KB, fits signed 24-bit), no loop-carried 64-bit adds.
// Per K-chunk (32): 8 b128 B-loads + 16 b128 A-loads -> 24 WMMAs (bf16x3).
// ---------------------------------------------------------------------------
__global__ __launch_bounds__(256) void cos_wmma_gemm(
    const __bf16* __restrict__ Shi, const __bf16* __restrict__ Slo,
    const __bf16* __restrict__ Qthi, const __bf16* __restrict__ Qtlo,
    const float* __restrict__ sn, const float* __restrict__ qn,
    float* __restrict__ out)
{
    const int tid   = threadIdx.x;
    const int lane  = tid & 31;
    const int w     = tid >> 5;            // wave id 0..7, all along M (n)
    const int mrow  = lane & 15;           // A: M row within tile / D: N col
    const int h     = lane >> 4;           // lane-half select

    const int n_wave  = blockIdx.x * 512 + w * 64;   // wave's first support row
    const int q_base0 = blockIdx.y * 32;             // WG's first query

    v8f acc[4][2] = {};

    // Clamp A row indices at the N edge (keeps EXEC all-ones for WMMA;
    // duplicated rows are discarded by guarded stores).
    int arow[4];
#pragma unroll
    for (int m = 0; m < 4; ++m) {
        int r = n_wave + m * 16 + mrow;
        arow[m] = (r < NS) ? r : (NS - 1);
    }

    // Loop-invariant base pointers; per-iteration offsets are compile-time.
    const __bf16* bbase_h = Qthi + (size_t)(q_base0 + mrow) * KD + h * 8;
    const __bf16* bbase_l = Qtlo + (size_t)(q_base0 + mrow) * KD + h * 8;
    const __bf16* abase_h[4];
    const __bf16* abase_l[4];
#pragma unroll
    for (int m = 0; m < 4; ++m) {
        abase_h[m] = Shi + (size_t)arow[m] * KD + h * 8;
        abase_l[m] = Slo + (size_t)arow[m] * KD + h * 8;
    }

#pragma unroll
    for (int kc = 0; kc < KD; kc += 32) {
        // ---- B operands: lane = q column, K in same lane-half split as A ----
        union { v16bf v; uint4 u[2]; } bhi[2], blo[2];
        const size_t boff = (size_t)kc;
#pragma unroll
        for (int nn = 0; nn < 2; ++nn) {
            const __bf16* bph = bbase_h + boff + (size_t)nn * 16 * KD;
            const __bf16* bpl = bbase_l + boff + (size_t)nn * 16 * KD;
            bhi[nn].u[0] = *(const uint4*)(bph);
            bhi[nn].u[1] = *(const uint4*)(bph + 16);
            blo[nn].u[0] = *(const uint4*)(bpl);
            blo[nn].u[1] = *(const uint4*)(bpl + 16);
        }

#pragma unroll
        for (int m = 0; m < 4; ++m) {
            union { v16bf v; uint4 u[2]; } ahi, alo;
            // ISA 16-bit A layout: lane-half h splits the 32-K chunk
            ahi.u[0] = *(const uint4*)(abase_h[m] + kc);
            ahi.u[1] = *(const uint4*)(abase_h[m] + kc + 16);
            alo.u[0] = *(const uint4*)(abase_l[m] + kc);
            alo.u[1] = *(const uint4*)(abase_l[m] + kc + 16);

#pragma unroll
            for (int nn = 0; nn < 2; ++nn) {
                // bf16x3 error-corrected product: hi*hi + hi*lo + lo*hi
                acc[m][nn] = __builtin_amdgcn_wmma_f32_16x16x32_bf16(
                    false, ahi.v, false, bhi[nn].v, (short)0, acc[m][nn], false, false);
                acc[m][nn] = __builtin_amdgcn_wmma_f32_16x16x32_bf16(
                    false, ahi.v, false, blo[nn].v, (short)0, acc[m][nn], false, false);
                acc[m][nn] = __builtin_amdgcn_wmma_f32_16x16x32_bf16(
                    false, alo.v, false, bhi[nn].v, (short)0, acc[m][nn], false, false);
            }
        }
    }

    // ---- Epilogue: D layout -> lane holds q = q_base+mrow, VGPR r holds
    //      n = n_tile + 8*h + r. Scale by 1/max(qn*sn, eps); NT stores so
    //      the streamed output does not evict the L2-resident operands. ----
#pragma unroll
    for (int nn = 0; nn < 2; ++nn) {
        const int   q   = q_base0 + nn * 16 + mrow;
        const float qnv = qn[q];
#pragma unroll
        for (int m = 0; m < 4; ++m) {
            const int n0 = n_wave + m * 16 + h * 8;   // 8 consecutive n
            if (n0 + 8 <= NS) {
                f4v s0 = *(const f4v*)(sn + n0);
                f4v s1 = *(const f4v*)(sn + n0 + 4);
                float sv[8] = {s0.x, s0.y, s0.z, s0.w, s1.x, s1.y, s1.z, s1.w};
                float ov[8];
#pragma unroll
                for (int r = 0; r < 8; ++r)
                    ov[r] = acc[m][nn][r] / fmaxf(qnv * sv[r], EPSV);
                float* op = out + (size_t)q * NS + n0;
                f4v o0 = {ov[0], ov[1], ov[2], ov[3]};
                f4v o1 = {ov[4], ov[5], ov[6], ov[7]};
                __builtin_nontemporal_store(o0, (f4v*)(op));
                __builtin_nontemporal_store(o1, (f4v*)(op + 4));
            } else {
#pragma unroll
                for (int r = 0; r < 8; ++r) {
                    const int n = n0 + r;
                    if (n < NS) {
                        float v = acc[m][nn][r] / fmaxf(qnv * sn[n], EPSV);
                        __builtin_nontemporal_store(v, out + (size_t)q * NS + n);
                    }
                }
            }
        }
    }
}

// ---------------------------------------------------------------------------
// Launch
// ---------------------------------------------------------------------------
extern "C" void kernel_launch(void* const* d_in, const int* in_sizes, int n_in,
                              void* d_out, int out_size, void* d_ws, size_t ws_size,
                              hipStream_t stream) {
    (void)in_sizes; (void)n_in; (void)out_size; (void)ws_size;

    const float* S = (const float*)d_in[0];   // support_set [50000, 512]
    const float* Q = (const float*)d_in[1];   // query_set   [2048, 512]
    float* out     = (float*)d_out;           // [2048, 50000]

    // Workspace layout (~107 MB)
    constexpr size_t S_BYTES  = (size_t)NS * KD * 2;   // 51,200,000
    constexpr size_t QT_BYTES = (size_t)KD * NQ * 2;   //  2,097,152
    constexpr size_t SN_BYTES = ((size_t)NS * 4 + 255) & ~(size_t)255;

    char* ws = (char*)d_ws;
    __bf16* Shi  = (__bf16*)(ws);
    __bf16* Slo  = (__bf16*)(ws + S_BYTES);
    __bf16* Qthi = (__bf16*)(ws + 2 * S_BYTES);
    __bf16* Qtlo = (__bf16*)(ws + 2 * S_BYTES + QT_BYTES);
    float*  sn   = (float*) (ws + 2 * S_BYTES + 2 * QT_BYTES);
    float*  qn   = (float*) (ws + 2 * S_BYTES + 2 * QT_BYTES + SN_BYTES);

    cos_prep_support<<<dim3(NS), dim3(256), 0, stream>>>(S, Shi, Slo, sn);
    cos_prep_query  <<<dim3(NQ), dim3(256), 0, stream>>>(Q, Qthi, Qtlo, qn);

    dim3 grid((NS + 511) / 512, NQ / 32);   // 98 x 64
    cos_wmma_gemm<<<grid, dim3(256), 0, stream>>>(Shi, Slo, Qthi, Qtlo, sn, qn, out);
}
